// TransformerAttentionwithPisition_34222299414553
// MI455X (gfx1250) — matrix operand order repaired and checked by reading, not compiled, not run on hardware
//
// MI455X / gfx1250: fused transformer-attention-with-position via algebraic
// contraction.  final[b] = sum_{q,k} softmax_q(scale*(X M X^T + Xu 1^T + 1 (Xw)^T + c))[q,k]
//                          * (X P + 1 g0^T)[k,q] + bfc
// Precompute: M = Wq Wk^T (bf16), P = Wv Wfc_r^T (bf16), u, w, g0, c.
// Main kernel: 1 workgroup = 4 waves = 4 samples; M tiles double-buffered into
// LDS with global_load_async_to_lds_b128; all GEMMs on v_wmma_f32_16x16x32_bf16.
// Pipeline per subtile: s_wait_asynccnt 0 -> s_barrier -> issue(s+1) -> compute(s).

#include <hip/hip_runtime.h>
#include <stdint.h>

#define IDIM 2048
#define TOK  16
#define NBATCH 2048
#define NB 4   // samples (waves) per workgroup

typedef __attribute__((ext_vector_type(16))) __bf16 v16bf;
typedef __attribute__((ext_vector_type(8)))  __bf16 v8bf;
typedef __attribute__((ext_vector_type(8)))  float  v8f;
typedef __attribute__((ext_vector_type(4)))  float  v4f;

// ---------------------------------------------------------------- utilities

static __device__ inline uint32_t lds_off(const void* p) {
  // generic LDS pointer: addr[31:0] is the LDS byte offset (ISA 10.2)
  return (uint32_t)(uintptr_t)p;
}

static __device__ inline void async_copy_b128(uint32_t lds_addr, const void* sbase,
                                              uint32_t voff) {
  asm volatile("global_load_async_to_lds_b128 %0, %1, %2"
               :: "v"(lds_addr), "v"(voff), "s"(sbase) : "memory");
}
static __device__ inline void wait_async_0() {
  asm volatile("s_wait_asynccnt 0x0" ::: "memory");
}

template <int XORMASK>
static __device__ inline float lane_xor_f(float v) {
  int r = __builtin_amdgcn_ds_swizzle(__float_as_int(v), (XORMASK << 10) | 0x1F);
  return __int_as_float(r);
}

// A fragment, 16-bit 16x32 (ISA 7.12.2): lanes 0-15 row m: K {0..7,16..23};
// lanes 16-31 row m: K {8..15,24..31}.
static __device__ inline v16bf load_a_frag(const __bf16* base, int rowStride,
                                           int k0, int lane) {
  const int m = lane & 15, h = lane >> 4;
  const __bf16* r = base + (size_t)m * rowStride + k0 + h * 8;
  v8bf lo = *(const v8bf*)(r);
  v8bf hi = *(const v8bf*)(r + 16);
  v16bf a;
#pragma unroll
  for (int i = 0; i < 8; ++i) { a[i] = lo[i]; a[8 + i] = hi[i]; }
  return a;
}

// B fragment, 16-bit 32x16: lanes 0-15 col n: K {0..7,16..23}; lanes 16-31: K {8..15,24..31}.
static __device__ inline v16bf load_b_frag(const __bf16* base, int rowStride,
                                           int k0, int lane) {
  const int n = lane & 15, h = lane >> 4;
  const __bf16* r = base + (size_t)n * rowStride + k0 + h * 8;
  v8bf lo = *(const v8bf*)(r);
  v8bf hi = *(const v8bf*)(r + 16);
  v16bf b;
#pragma unroll
  for (int i = 0; i < 8; ++i) { b[i] = lo[i]; b[8 + i] = hi[i]; }
  return b;
}

static __device__ inline v8f wmma_bf16(v16bf a, v16bf b, v8f c) {
  return __builtin_amdgcn_wmma_f32_16x16x32_bf16(false, a, false, b,
                                                 (short)0, c, false, false);
}

// ---------------------------------------------------- precompute vectors / P

__global__ __launch_bounds__(256) void k_vec(
    const float* __restrict__ Wq, const float* __restrict__ bq,
    const float* __restrict__ Wk, const float* __restrict__ bk,
    const float* __restrict__ Wv, const float* __restrict__ bv,
    const float* __restrict__ Wfc,
    float* __restrict__ u, float* __restrict__ w,
    __bf16* __restrict__ Pt, float* __restrict__ g0c) {
  __shared__ float red[256];
  const int d = blockIdx.x, tid = threadIdx.x;
  float pu = 0.f, pw = 0.f, pj[16];
#pragma unroll
  for (int j = 0; j < 16; ++j) pj[j] = 0.f;
  for (int e = tid; e < IDIM; e += 256) {
    const float wq = Wq[(size_t)d * IDIM + e];
    const float wk = Wk[(size_t)d * IDIM + e];
    const float wvv = Wv[(size_t)d * IDIM + e];
    pu += wq * bk[e];
    pw += wk * bq[e];
#pragma unroll
    for (int j = 0; j < 16; ++j) pj[j] += wvv * Wfc[(size_t)j * IDIM + e];
  }
  auto reduce = [&](float v) -> float {
    __syncthreads();
    red[tid] = v;
    __syncthreads();
    for (int off = 128; off; off >>= 1) {
      if (tid < off) red[tid] += red[tid + off];
      __syncthreads();
    }
    return red[0];
  };
  float su = reduce(pu);
  float sw = reduce(pw);
  if (tid == 0) { u[d] = su; w[d] = sw; }
  for (int j = 0; j < 16; ++j) {
    float s = reduce(pj[j]);
    if (tid == 0) Pt[(size_t)j * IDIM + d] = (__bf16)s;
  }
  if (d == 0) {  // scalars c and g0 (block-uniform branch)
    float pc = 0.f, pg[16];
#pragma unroll
    for (int j = 0; j < 16; ++j) pg[j] = 0.f;
    for (int e = tid; e < IDIM; e += 256) {
      pc += bq[e] * bk[e];
      const float bve = bv[e];
#pragma unroll
      for (int j = 0; j < 16; ++j) pg[j] += bve * Wfc[(size_t)j * IDIM + e];
    }
    float sc = reduce(pc);
    if (tid == 0) g0c[16] = sc;
    for (int j = 0; j < 16; ++j) {
      float s = reduce(pg[j]);
      if (tid == 0) g0c[j] = s;
    }
  }
}

// ------------------------------------------- precompute  Mt[dp][d]=Wk·Wq^T

__global__ __launch_bounds__(256) void k_gemm_m(const float* __restrict__ Wk,
                                                const float* __restrict__ Wq,
                                                __bf16* __restrict__ Mt) {
  __shared__ alignas(16) __bf16 Ap[128][40];  // Wk rows panel (e-tile 32, pad 8)
  __shared__ alignas(16) __bf16 Bp[128][40];  // Wq rows panel
  const int tid = threadIdx.x, lane = tid & 31, w = tid >> 5;
  const int dp0 = blockIdx.y * 128, d0 = blockIdx.x * 128;
  const int tr = (w >> 1) * 32, tc = (w & 1) * 64;  // per-wave 32x64 sub-tile
  v8f acc[2][4] = {};
  const int rrow = tid >> 1, seg = tid & 1;
  for (int et = 0; et < 64; ++et) {
    const int e0 = et * 32 + seg * 16;
    v4f a4[4], b4[4];
#pragma unroll
    for (int j = 0; j < 4; ++j) {
      a4[j] = *(const v4f*)(Wk + (size_t)(dp0 + rrow) * IDIM + e0 + j * 4);
      b4[j] = *(const v4f*)(Wq + (size_t)(d0 + rrow) * IDIM + e0 + j * 4);
    }
    __syncthreads();
#pragma unroll
    for (int j = 0; j < 4; ++j)
#pragma unroll
      for (int q = 0; q < 4; ++q) {
        Ap[rrow][seg * 16 + j * 4 + q] = (__bf16)a4[j][q];
        Bp[rrow][seg * 16 + j * 4 + q] = (__bf16)b4[j][q];
      }
    __syncthreads();
    v16bf a0 = load_a_frag(&Ap[tr][0], 40, 0, lane);
    v16bf a1 = load_a_frag(&Ap[tr + 16][0], 40, 0, lane);
#pragma unroll
    for (int c4 = 0; c4 < 4; ++c4) {
      v16bf bb = load_b_frag(&Bp[tc + c4 * 16][0], 40, 0, lane);
      acc[0][c4] = wmma_bf16(a0, bb, acc[0][c4]);
      acc[1][c4] = wmma_bf16(a1, bb, acc[1][c4]);
    }
  }
#pragma unroll
  for (int i = 0; i < 2; ++i)
#pragma unroll
    for (int c4 = 0; c4 < 4; ++c4)
#pragma unroll
      for (int r = 0; r < 8; ++r) {
        const int row = dp0 + tr + i * 16 + r + ((lane >> 4) << 3);
        const int col = d0 + tc + c4 * 16 + (lane & 15);
        Mt[(size_t)row * IDIM + col] = (__bf16)acc[i][c4][r];
      }
}

// --------------------------------------------------------- fused main kernel

__global__ __launch_bounds__(NB * 32) void k_fused(
    const float* __restrict__ x, const float* __restrict__ pos,
    const __bf16* __restrict__ Mt, const __bf16* __restrict__ Pt,
    const float* __restrict__ u, const float* __restrict__ wv,
    const float* __restrict__ g0c, const float* __restrict__ bfc,
    float* __restrict__ out) {
  __shared__ alignas(16) __bf16 Xs[NB][TOK][IDIM + 8];   // 263168 B
  __shared__ alignas(16) __bf16 Mtile[2][32][264];       //  33792 B
  __shared__ alignas(16) __bf16 Ys[NB][16][40];          //   5120 B
  __shared__ float SdG[NB][512];                         //   8192 B
  __shared__ float Red[NB][32];                          //    512 B
  const int lane = threadIdx.x & 31, w = threadIdx.x >> 5;
  const int b = blockIdx.x * NB + w;
  const v8f vzero = {};

  // preissue M subtile 0 (32 d'-rows x 256 d): 8 rows per wave, b128 per lane;
  // the DMA overlaps the X staging loop below.
  {
    const uint32_t l0 = lds_off(&Mtile[0][w * 8][0]);
    const __bf16* g = Mt + (size_t)(w * 8) * IDIM;
#pragma unroll
    for (int r = 0; r < 8; ++r)
      async_copy_b128(l0 + r * 264 * 2 + lane * 16, g + (size_t)r * IDIM,
                      (uint32_t)(lane * 16));
  }

  // stage X[b] (+pos) into LDS as bf16 (wave-private rows)
  for (int i = lane; i < TOK * IDIM / 4; i += 32) {
    const int t = i >> 9;
    const int d4 = (i & 511) << 2;
    v4f xv = *(const v4f*)(x + ((size_t)b * TOK + t) * IDIM + d4);
    v4f pv = *(const v4f*)(pos + (size_t)t * IDIM + d4);
#pragma unroll
    for (int j = 0; j < 4; ++j) Xs[w][t][d4 + j] = (__bf16)(xv[j] + pv[j]);
  }

  v8f acc0 = vzero, acc1 = vzero, attacc = vzero;

  const int S = 64 * 8;  // 64 d'-chunks x 8 K-subtiles
  for (int s = 0; s < S; ++s) {
    const int jc = s >> 3, kt = s & 7, buf = s & 1;
    // my 8 rows of subtile s are in LDS:
    wait_async_0();
    // => everyone's rows of subtile s are in LDS, and everyone finished
    //    compute(s-1), so buf^1 may be overwritten:
    __syncthreads();
    if (s + 1 < S) {
      const int jc2 = (s + 1) >> 3, kt2 = (s + 1) & 7;
      const uint32_t l0 = lds_off(&Mtile[buf ^ 1][w * 8][0]);
      const __bf16* g = Mt + (size_t)(jc2 * 32 + w * 8) * IDIM + kt2 * 256;
#pragma unroll
      for (int r = 0; r < 8; ++r)
        async_copy_b128(l0 + r * 264 * 2 + lane * 16, g + (size_t)r * IDIM,
                        (uint32_t)(lane * 16));
    }
    const int dbase = kt * 256;
#pragma unroll
    for (int ks = 0; ks < 8; ++ks) {
      v16bf a  = load_a_frag(&Xs[w][0][0], IDIM + 8, dbase + ks * 32, lane);
      v16bf b0 = load_b_frag(&Mtile[buf][0][0], 264, ks * 32, lane);
      v16bf b1 = load_b_frag(&Mtile[buf][16][0], 264, ks * 32, lane);
      acc0 = wmma_bf16(a, b0, acc0);
      acc1 = wmma_bf16(a, b1, acc1);
    }
    if (kt == 7) {  // chunk done: att += Y_chunk · X_chunk^T  (K=32)
#pragma unroll
      for (int r = 0; r < 8; ++r) {
        const int q = r + ((lane >> 4) << 3);
        Ys[w][q][lane & 15]        = (__bf16)acc0[r];
        Ys[w][q][16 + (lane & 15)] = (__bf16)acc1[r];
      }
      v16bf aY = load_a_frag(&Ys[w][0][0], 40, 0, lane);
      v16bf bX = load_b_frag(&Xs[w][0][0], IDIM + 8, jc * 32, lane);
      attacc = wmma_bf16(aY, bX, attacc);
      acc0 = vzero;
      acc1 = vzero;
    }
  }

  // G = X · P   (B-fragments straight from global bf16 Pt[16][2048])
  v8f gacc = vzero;
  for (int kk = 0; kk < 64; ++kk) {
    v16bf a  = load_a_frag(&Xs[w][0][0], IDIM + 8, kk * 32, lane);
    v16bf bP = load_b_frag(Pt, IDIM, kk * 32, lane);
    gacc = wmma_bf16(a, bP, gacc);
  }

  // Xu[q], Xw[q]: lane pair (q, q+16-halves of d)
  {
    const int q = lane & 15, h = lane >> 4;
    float pu = 0.f, pw = 0.f;
    for (int i = 0; i < 1024; i += 4) {
      const int d = h * 1024 + i;
      v4f uv = *(const v4f*)(u + d);
      v4f w4 = *(const v4f*)(wv + d);
#pragma unroll
      for (int j = 0; j < 4; ++j) {
        const float xb = (float)Xs[w][q][d + j];
        pu += xb * uv[j];
        pw += xb * w4[j];
      }
    }
    pu += lane_xor_f<16>(pu);
    pw += lane_xor_f<16>(pw);
    if (h == 0) { Red[w][q] = pu; Red[w][16 + q] = pw; }
  }

  // att finalize + softmax over q (axis=1): column k lives in lanes {k, k+16}
  const float cterm = g0c[16];
  const float scale = 0.022097086912079608f;  // 1/sqrt(2048)
  const int k = lane & 15;
  const float xwk = Red[w][16 + k];
  float vals[8];
#pragma unroll
  for (int r = 0; r < 8; ++r) {
    const int q = r + ((lane >> 4) << 3);
    vals[r] = (attacc[r] + Red[w][q] + xwk + cterm) * scale;
  }
  float m = vals[0];
#pragma unroll
  for (int r = 1; r < 8; ++r) m = fmaxf(m, vals[r]);
  m = fmaxf(m, lane_xor_f<16>(m));
  float ex[8], es = 0.f;
#pragma unroll
  for (int r = 0; r < 8; ++r) { ex[r] = __expf(vals[r] - m); es += ex[r]; }
  es += lane_xor_f<16>(es);
  const float inv = 1.0f / es;

  float* Sd = &SdG[w][0];
  float* Gd = &SdG[w][256];
  const float g0q = g0c[lane & 15];
#pragma unroll
  for (int r = 0; r < 8; ++r) {
    const int q = r + ((lane >> 4) << 3);
    Sd[q * 16 + k] = ex[r] * inv;                                   // S[q,k]
    Gd[(lane & 15) * 16 + r + ((lane >> 4) << 3)] = gacc[r] + g0q;  // G'[k,q] at [q*16+k]
  }

  // final = sum S[q,k] * G'[k,q] + bfc
  float p = 0.f;
#pragma unroll
  for (int i = 0; i < 8; ++i) p += Sd[lane * 8 + i] * Gd[lane * 8 + i];
  p += lane_xor_f<1>(p);
  p += lane_xor_f<2>(p);
  p += lane_xor_f<4>(p);
  p += lane_xor_f<8>(p);
  p += lane_xor_f<16>(p);
  if (lane == 0) out[b] = p + bfc[0];
}

// ------------------------------------------------------------------- launch

extern "C" void kernel_launch(void* const* d_in, const int* in_sizes, int n_in,
                              void* d_out, int out_size, void* d_ws, size_t ws_size,
                              hipStream_t stream) {
  const float* x   = (const float*)d_in[0];
  const float* Wq  = (const float*)d_in[1];
  const float* bq  = (const float*)d_in[2];
  const float* Wk  = (const float*)d_in[3];
  const float* bk  = (const float*)d_in[4];
  const float* Wv  = (const float*)d_in[5];
  const float* bv  = (const float*)d_in[6];
  const float* Wfc = (const float*)d_in[7];
  const float* bfc = (const float*)d_in[8];
  const float* pos = (const float*)d_in[9];

  char* ws = (char*)d_ws;
  __bf16* Mt  = (__bf16*)ws;                                   // 8 MB
  __bf16* Pt  = (__bf16*)(ws + (size_t)8 * 1024 * 1024);       // 64 KB
  float*  u   = (float*)(ws + (size_t)8 * 1024 * 1024 + 65536);
  float*  w   = u + IDIM;
  float*  g0c = w + IDIM;  // g0[16] + c
  const size_t need = (size_t)8 * 1024 * 1024 + 65536 + sizeof(float) * (IDIM + IDIM + 17);
  if (ws_size < need) return;

  k_vec<<<dim3(IDIM), dim3(256), 0, stream>>>(Wq, bq, Wk, bk, Wv, bv, Wfc,
                                              u, w, Pt, g0c);
  k_gemm_m<<<dim3(16, 16), dim3(256), 0, stream>>>(Wk, Wq, Mt);
  k_fused<<<dim3(NBATCH / NB), dim3(NB * 32), 0, stream>>>(
      x, pos, Mt, Pt, u, w, g0c, bfc, (float*)d_out);
}
